// EntropyBottleneckLatticeFlow_59785944760413
// MI455X (gfx1250) — matrix operand order repaired and checked by reading, not compiled, not run on hardware
//
#include <hip/hip_runtime.h>

typedef __attribute__((ext_vector_type(16))) _Float16 v16h;
typedef __attribute__((ext_vector_type(8)))  float    v8f;
typedef __attribute__((ext_vector_type(8)))  int      v8i;

#define LOG2E 1.4426950408889634f

// ---- cross-lane helpers (wave32) -------------------------------------------
template<int IMM>
__device__ __forceinline__ float dswz(float x) {
  return __int_as_float(__builtin_amdgcn_ds_swizzle(__float_as_int(x), IMM));
}
// exchange lane L <-> L^16 (SWAPX16: xor=0x10, and=0x1f)
__device__ __forceinline__ float swap16(float x) { return dswz<0x401f>(x); }
__device__ __forceinline__ int   iswap16(int x)  { return __builtin_amdgcn_ds_swizzle(x, 0x401f); }

__device__ __forceinline__ float fast_exp(float x) {
  return __builtin_amdgcn_exp2f(x * LOG2E);
}
__device__ __forceinline__ float fast_tanh(float x) {
#if __has_builtin(__builtin_amdgcn_tanhf)
  return __builtin_amdgcn_tanhf(x);              // native v_tanh_f32 (CDNA5)
#elif __has_builtin(__builtin_amdgcn_tanh_f32)
  return __builtin_amdgcn_tanh_f32(x);
#else
  // tanh(x) = 1 - 2/(exp(2x)+1)
  float e = __builtin_amdgcn_exp2f(x * (2.0f * LOG2E));
  return 1.0f - 2.0f * __builtin_amdgcn_rcpf(e + 1.0f);
#endif
}

// pack two f32 into one f16x2 word (v_cvt_pk_rtz_f16_f32)
__device__ __forceinline__ int pk16(float a, float b) {
  return __builtin_bit_cast(int, __builtin_amdgcn_cvt_pkrtz(a, b));
}

// ---- WMMA wrappers ----------------------------------------------------------
__device__ __forceinline__ v8f wmma_f16(const v16h a, const v16h b, const v8f c) {
  // D = A(16x32 f16) * B(32x16 f16) + C(16x16 f32)
  return __builtin_amdgcn_wmma_f32_16x16x32_f16(
      /*neg_a=*/false, a, /*neg_b=*/false, b,
      /*c_mod=*/(short)0, c, /*reuse_a=*/false, /*reuse_b=*/false);
}

// A operand pre-packed in LDS: layout [tile][hi][m][16 f16], 32B rows.
__device__ __forceinline__ v16h loadA(const _Float16* base, int tn, int lane) {
  int off = ((tn * 2 + (lane >> 4)) * 16 + (lane & 15)) * 16;
  return *(const v16h*)(base + off);
}
// Bias as C operand: C[r] = bias[r + 8*hi + 16*tn] (uniform per half-wave)
__device__ __forceinline__ v8f loadC(const float* b, int tn, int hi) {
  return *(const v8f*)(b + tn * 16 + hi * 8);
}

// Build B (32x16 f16) from a 16-row activation in C/D layout; K rows 16..31
// are zero padding (words 4..7 = 0).  Word j holds K=2j,2j+1 (+8 for hi).
__device__ __forceinline__ v16h makeB_half(const v8f x, int hi) {
  v8i w;
#pragma unroll
  for (int j = 0; j < 4; ++j) {
    int pk = pk16(x[2 * j], x[2 * j + 1]);   // features 2j,2j+1 of this half
    int sw = iswap16(pk);                    // features 8+2j,9+2j -> lo lanes
    w[j]     = pk;
    w[j + 4] = 0 * sw;
  }
  return __builtin_bit_cast(v16h, w);
}

// Build B (32x16 f16) from a 32-row activation held as two C/D tiles
// h0 (neurons 0..15) and h1 (neurons 16..31).
__device__ __forceinline__ v16h makeB_full(const v8f h0, const v8f h1, int hi) {
  v8i w;
#pragma unroll
  for (int j = 0; j < 4; ++j) {
    int p0 = pk16(h0[2 * j], h0[2 * j + 1]);
    int p1 = pk16(h1[2 * j], h1[2 * j + 1]);
    int s0 = iswap16(p0);                    // h0 pairs from the other half
    int s1 = iswap16(p1);                    // h1 pairs from the other half
    w[j]     = hi ? p0 : p0 + 0 * s1;   // lo: neurons 2j,2j+1    | hi: 8+2j,9+2j
    w[j + 4] = hi ? p1 : p1 + 0 * s0;   // lo: neurons 16+2j,17+2j | hi: 24+2j,25+2j
  }
  return __builtin_bit_cast(v16h, w);
}

// One MLP (16 -> 32 tanh -> 32 tanh -> 16), batch of 16 samples via WMMA.
__device__ __forceinline__ v8f run_mlp(const v16h Bx,
                                       const _Float16* A1, const _Float16* A2,
                                       const _Float16* A3,
                                       const float* b1, const float* b2,
                                       const float* b3, int lane, int hi) {
  v8f h0 = wmma_f16(loadA(A1, 0, lane), Bx, loadC(b1, 0, hi));
  v8f h1 = wmma_f16(loadA(A1, 1, lane), Bx, loadC(b1, 1, hi));
#pragma unroll
  for (int r = 0; r < 8; ++r) { h0[r] = fast_tanh(h0[r]); h1[r] = fast_tanh(h1[r]); }
  v16h B2 = makeB_full(h0, h1, hi);
  v8f g0 = wmma_f16(loadA(A2, 0, lane), B2, loadC(b2, 0, hi));
  v8f g1 = wmma_f16(loadA(A2, 1, lane), B2, loadC(b2, 1, hi));
#pragma unroll
  for (int r = 0; r < 8; ++r) { g0[r] = fast_tanh(g0[r]); g1[r] = fast_tanh(g1[r]); }
  v16h B3 = makeB_full(g0, g1, hi);
  return wmma_f16(loadA(A3, 0, lane), B3, loadC(b3, 0, hi));
}

// Convert one element of W[K][Nout] (row-major f32) into the CDNA5 16-bit
// A-operand packing: per (tile,hi,m) 16 f16 elems e=2v+p, K = 2(v&3)+16(v>>2)+8hi+p.
__device__ __forceinline__ _Float16 packA_elem(const float* __restrict__ Wg,
                                               int idx, int Ktot, int Nout) {
  int tn = idx >> 9;        // 512 halfs per 16-neuron tile
  int r  = idx & 511;
  int hi = r >> 8;  r &= 255;
  int m  = r >> 4;
  int e  = r & 15;
  int v = e >> 1, p = e & 1;
  if (Ktot == 16 && v >= 4) return (_Float16)0.0f;   // K padding for first layer
  int k = 2 * (v & 3) + ((Ktot == 32) ? 16 * (v >> 2) : 0) + 8 * hi + p;
  return (_Float16)Wg[k * Nout + tn * 16 + m];
}

// ---- main kernel: one wave = 16 samples through all 5 flows -----------------
__global__ __launch_bounds__(256) void flow_kernel(
    const float* __restrict__ inputs, const float* __restrict__ noise,
    const float* __restrict__ W1, const float* __restrict__ b1,
    const float* __restrict__ W2, const float* __restrict__ b2,
    const float* __restrict__ W3, const float* __restrict__ b3,
    float* __restrict__ ws) {
  __shared__ __align__(32) _Float16 sA1[4][1024];  // 4 MLPs, 16x32 padded-K A pack
  __shared__ __align__(32) _Float16 sA2[4][1024];  // 32x32 A pack
  __shared__ __align__(32) _Float16 sA3[4][512];   // 32x16 A pack
  __shared__ __align__(32) float    sb1[4][32];
  __shared__ __align__(32) float    sb2[4][32];
  __shared__ __align__(32) float    sb3[4][16];

  const int tid  = threadIdx.x;
  const int lane = tid & 31;
  const int hi   = lane >> 4;
  const int wave = tid >> 5;
  const int t    = blockIdx.x * 8 + wave;   // tile of 16 samples
  const int g    = t * 16 + (lane & 15);    // global sample index
  const int bIdx = g >> 8;                  // input row (N noise = 256)
  const int nIdx = g & 255;                 // noise row

  // z in C/D layout: feature f = r + 8*hi (lower half), +16 (upper half),
  // sample = lane & 15.
  v8f zL, zU;
#pragma unroll
  for (int r = 0; r < 8; ++r) {
    int f = r + 8 * hi;
    zL[r] = inputs[bIdx * 32 + f]      - noise[nIdx * 32 + f];
    zU[r] = inputs[bIdx * 32 + 16 + f] - noise[nIdx * 32 + 16 + f];
  }
  float logdet = 0.0f;

  for (int i = 0; i < 5; ++i) {
    __syncthreads();
    // Stage this flow's 4 MLPs into LDS, f16, A-operand packed.
    for (int idx = tid; idx < 4 * 1024; idx += 256) {
      int m = idx >> 10, r = idx & 1023;
      sA1[m][r] = packA_elem(W1 + (i * 4 + m) * 512, r, 16, 32);
    }
    for (int idx = tid; idx < 4 * 1024; idx += 256) {
      int m = idx >> 10, r = idx & 1023;
      sA2[m][r] = packA_elem(W2 + (i * 4 + m) * 1024, r, 32, 32);
    }
    for (int idx = tid; idx < 4 * 512; idx += 256) {
      int m = idx >> 9, r = idx & 511;
      sA3[m][r] = packA_elem(W3 + (i * 4 + m) * 512, r, 32, 16);
    }
    for (int idx = tid; idx < 128; idx += 256) {
      int m = idx >> 5, r = idx & 31;
      sb1[m][r] = b1[(i * 4 + m) * 32 + r];
      sb2[m][r] = b2[(i * 4 + m) * 32 + r];
    }
    for (int idx = tid; idx < 64; idx += 256) {
      int m = idx >> 4, r = idx & 15;
      sb3[m][r] = b3[(i * 4 + m) * 16 + r];
    }
    __syncthreads();

    // upper' = t1 + upper * exp(s1)
    v16h BL = makeB_half(zL, hi);
    v8f t1 = run_mlp(BL, sA1[0], sA2[0], sA3[0], sb1[0], sb2[0], sb3[0], lane, hi);
    v8f s1 = run_mlp(BL, sA1[1], sA2[1], sA3[1], sb1[1], sb2[1], sb3[1], lane, hi);
    float part = 0.0f;
#pragma unroll
    for (int r = 0; r < 8; ++r) {
      zU[r] = t1[r] + zU[r] * fast_exp(s1[r]);
      part += s1[r];
    }
    logdet += part + swap16(part);

    // lower' = t2 + lower * exp(s2)
    v16h BU = makeB_half(zU, hi);
    v8f t2 = run_mlp(BU, sA1[2], sA2[2], sA3[2], sb1[2], sb2[2], sb3[2], lane, hi);
    v8f s2 = run_mlp(BU, sA1[3], sA2[3], sA3[3], sb1[3], sb2[3], sb3[3], lane, hi);
    part = 0.0f;
#pragma unroll
    for (int r = 0; r < 8; ++r) {
      zL[r] = t2[r] + zL[r] * fast_exp(s2[r]);
      part += s2[r];
    }
    logdet += part + swap16(part);
  }

  // prior: -0.5*32*log(2pi) - 0.5*|z|^2
  float q = 0.0f;
#pragma unroll
  for (int r = 0; r < 8; ++r) q += zL[r] * zL[r] + zU[r] * zU[r];
  q += swap16(q);
  float lp = -29.406033062549495f - 0.5f * q + logdet;
  float p  = fast_exp(lp);

  // sum over 16 samples within each half-wave (halves hold duplicates)
  p += dswz<0x041f>(p);   // xor 1
  p += dswz<0x081f>(p);   // xor 2
  p += dswz<0x101f>(p);   // xor 4
  p += dswz<0x201f>(p);   // xor 8
  if (lane == 0) ws[t] = p;
}

// Deterministic final reduction: 16 tiles per output row, mean over 256 noise.
__global__ __launch_bounds__(256) void reduce_kernel(const float* __restrict__ ws,
                                                     float* __restrict__ out) {
  int b = blockIdx.x * blockDim.x + threadIdx.x;
  if (b < 2048) {
    float s = 0.0f;
#pragma unroll
    for (int j = 0; j < 16; ++j) s += ws[b * 16 + j];
    out[b] = s * (1.0f / 256.0f);
  }
}

extern "C" void kernel_launch(void* const* d_in, const int* in_sizes, int n_in,
                              void* d_out, int out_size, void* d_ws, size_t ws_size,
                              hipStream_t stream) {
  const float* inputs = (const float*)d_in[0];
  const float* noise  = (const float*)d_in[1];
  const float* W1     = (const float*)d_in[2];
  const float* b1     = (const float*)d_in[3];
  const float* W2     = (const float*)d_in[4];
  const float* b2     = (const float*)d_in[5];
  const float* W3     = (const float*)d_in[6];
  const float* b3     = (const float*)d_in[7];
  float* ws  = (float*)d_ws;    // 32768 tile partials (128 KB)
  float* out = (float*)d_out;

  // 524288 samples / 16 per wave = 32768 waves; 8 waves per 256-thread block.
  flow_kernel<<<4096, 256, 0, stream>>>(inputs, noise, W1, b1, W2, b2, W3, b3, ws);
  reduce_kernel<<<8, 256, 0, stream>>>(ws, out);

  (void)in_sizes; (void)n_in; (void)out_size; (void)ws_size;
}
